// CustomGNN_80032420594210
// MI455X (gfx1250) — matrix operand order repaired and checked by reading, not compiled, not run on hardware
//
#include <hip/hip_runtime.h>
#include <hip/hip_bf16.h>
#include <cmath>

#define NN 50000
#define NE 800000

typedef __attribute__((ext_vector_type(16))) __bf16       v16bf;
typedef __attribute__((ext_vector_type(8)))  float        v8f;
typedef __attribute__((ext_vector_type(8)))  unsigned int v8u;

// ---------------------------------------------------------------- helpers
__device__ __forceinline__ unsigned short f2bf(float f) {
  unsigned int u = __float_as_uint(f);
  u += 0x7FFFu + ((u >> 16) & 1u);          // round-to-nearest-even
  return (unsigned short)(u >> 16);
}
// monotone float -> uint mapping so integer atomicMax == float max
__device__ __forceinline__ unsigned int orderF32(float f) {
  unsigned int u = __float_as_uint(f);
  return (u & 0x80000000u) ? ~u : (u | 0x80000000u);
}
__device__ __forceinline__ float unorderF32(unsigned int u) {
  unsigned int b = (u & 0x80000000u) ? (u ^ 0x80000000u) : ~u;
  return __uint_as_float(b);
}

// ------------------------------------------------- input prep kernels
__global__ void concat_cast_kernel(const float* __restrict__ feat,
                                   const float* __restrict__ pe,
                                   unsigned short* __restrict__ XB,
                                   int n, int inDim, int peDim) {
  int i = blockIdx.x * blockDim.x + threadIdx.x;
  int D = inDim + peDim;
  if (i >= n * D) return;
  int node = i / D, col = i - node * D;
  float v = (col < inDim) ? feat[(size_t)node * inDim + col]
                          : pe[(size_t)node * peDim + (col - inDim)];
  XB[i] = f2bf(v);
}

__global__ void cast_bf16_kernel(const float* __restrict__ in,
                                 unsigned short* __restrict__ out, int count) {
  int i = blockIdx.x * blockDim.x + threadIdx.x;
  if (i < count) out[i] = f2bf(in[i]);
}

// ------------------------------------------------- WMMA GEMM: C = A(bf16) @ B(bf16) + bias
// A: [M,K] bf16 row-major, B: [K,NOUT] bf16 row-major, C: [M,NOUT] f32.
// One wave computes a 16x64 strip (4 accumulators). All 4 B fragments are
// issued before the first WMMA of a k-step so the matrix pipe overlaps with
// the remaining loads (partial s_wait_loadcnt). K/NOUT compile-time so all
// strides strength-reduce to immediates. M%16==0, K%32==0, NOUT%64==0.
template <int K, int NOUT>
__global__ __launch_bounds__(256) void gemm_bf16_wmma_kernel(
    const unsigned short* __restrict__ A,
    const unsigned short* __restrict__ B,
    const float* __restrict__ bias,
    float* __restrict__ C,
    int M) {
  const int lane   = threadIdx.x & 31;
  const int wave   = blockIdx.x * (blockDim.x >> 5) + (threadIdx.x >> 5);
  const int nQuads = NOUT >> 6;                 // 64-wide column strips
  const int mTiles = M >> 4;
  if (wave >= mTiles * nQuads) return;          // wave-uniform: EXEC stays all-1s
  const int mTile = wave / nQuads;
  const int quad  = wave - mTile * nQuads;
  const int half  = lane >> 4;                  // 0: lanes 0-15, 1: lanes 16-31
  const int mrow  = (mTile << 4) + (lane & 15); // A-matrix: lane carries row M
  const int ncol0 = quad << 6;

  const unsigned short* Arow = A + (size_t)mrow * K;
  const unsigned short* Brow = B + (size_t)(lane & 15) + ncol0;  // + k0*NOUT per step
  v8f acc[4] = {};

#pragma unroll 2
  for (int k0 = 0; k0 < K; k0 += 32) {
    // A fragment: VGPR v holds K-pairs {0,2,4,6}|{16,18,20,22} (+8 for upper half-wave)
    v8u au;
#pragma unroll
    for (int v = 0; v < 8; ++v) {
      int koff = ((v < 4) ? (2 * v) : (16 + 2 * (v - 4))) + (half << 3);
      au[v] = *(const unsigned int*)(Arow + k0 + koff);
    }
    // Issue ALL B fragments before any WMMA so loads stay in flight while the
    // matrix pipe drains them in order. Lane carries column N; VGPR v holds
    // the same K-pair as A; 4 adjacent 16-wide N-tiles share this A fragment.
    const unsigned short* Bk = Brow + (size_t)k0 * NOUT;
    v8u bu[4];
#pragma unroll
    for (int t = 0; t < 4; ++t)
#pragma unroll
      for (int v = 0; v < 8; ++v)
        bu[t][v] = (unsigned int)Bk[(size_t)(((v < 4) ? (2 * v) : (16 + 2 * (v - 4))) + (half << 3)) * NOUT + 16 * t] | ((unsigned int)Bk[(size_t)(((v < 4) ? (2 * v) : (16 + 2 * (v - 4))) + (half << 3) + 1) * NOUT + 16 * t] << 16);

    v16bf a = __builtin_bit_cast(v16bf, au);
#pragma unroll
    for (int t = 0; t < 4; ++t) {
      v16bf b = __builtin_bit_cast(v16bf, bu[t]);
      acc[t] = __builtin_amdgcn_wmma_f32_16x16x32_bf16(false, a, false, b,
                                                       (short)0, acc[t], false, false);
    }
  }

  // C/D layout: VGPR r -> M = r (+8 for upper half-wave), N = lane&15
  const int nlane = lane & 15;
#pragma unroll
  for (int t = 0; t < 4; ++t) {
    const int nc = ncol0 + 16 * t + nlane;
    const float bval = bias[nc];
#pragma unroll
    for (int r = 0; r < 8; ++r) {
      int mr = (mTile << 4) + r + (half << 3);
      C[(size_t)mr * NOUT + nc] = acc[t][r] + bval;
    }
  }
}

// ------------------------------------------------- attention kernels
__global__ void init_nh_kernel(unsigned int* __restrict__ MMAX,
                               float* __restrict__ DEN, int count) {
  int i = blockIdx.x * blockDim.x + threadIdx.x;
  if (i < count) { MMAX[i] = 0u; DEN[i] = 0.0f; }
}

// Wave-per-edge: lanes cover contiguous channels (coalesced 128B requests),
// cross-lane reduction per head, lane 0 commits logit + ordered atomicMax.
__global__ __launch_bounds__(256) void edge_logits_max_kernel(
    const long long* __restrict__ ei,
    const float* __restrict__ Q,
    const float* __restrict__ Kb,
    float* __restrict__ LOG,
    unsigned int* __restrict__ MMAX,
    int E, int H, int Cch, int D, float scale) {
  const int lane = threadIdx.x & 31;
  const int e    = blockIdx.x * (blockDim.x >> 5) + (threadIdx.x >> 5);
  if (e >= E) return;
  const int s = (int)ei[e], d = (int)ei[E + e];
  const float* qrow = Q + (size_t)d * D;
  const float* krow = Kb + (size_t)s * D;
  for (int h = 0; h < H; ++h) {
    float part = 0.0f;
    for (int c0 = 0; c0 < Cch; c0 += 32) {
      int c = h * Cch + c0 + lane;
      part = fmaf(qrow[c], krow[c], part);
    }
#pragma unroll
    for (int off = 16; off; off >>= 1)
      part += __shfl_xor(part, off, 32);
    if (lane == 0) {
      float lg = part * scale;
      LOG[(size_t)e * H + h] = lg;
      atomicMax(&MMAX[(size_t)d * H + h], orderF32(lg));
    }
  }
}

__global__ void edge_exp_sum_kernel(const long long* __restrict__ ei,
                                    const float* __restrict__ LOG,
                                    const unsigned int* __restrict__ MMAX,
                                    float* __restrict__ EXPA,
                                    float* __restrict__ DEN, int E, int H) {
  int i = blockIdx.x * blockDim.x + threadIdx.x;
  if (i >= E * H) return;
  int e = i / H, h = i - e * H;
  int d = (int)ei[E + e];
  float m = unorderF32(MMAX[(size_t)d * H + h]);
  float a = __expf(LOG[i] - m);
  EXPA[i] = a;
  atomicAdd(&DEN[(size_t)d * H + h], a);
}

// thread per (edge, 4 channels): float4 value load + 4 scatter atomics
__global__ void edge_aggregate_kernel(const long long* __restrict__ ei,
                                      const float* __restrict__ Vb,
                                      const float* __restrict__ EXPA,
                                      const float* __restrict__ DEN,
                                      float* __restrict__ OUTACC,
                                      int E, int H, int Cch, int D) {
  const int D4 = D >> 2;
  long long i = (long long)blockIdx.x * blockDim.x + threadIdx.x;
  if (i >= (long long)E * D4) return;
  int e  = (int)(i / D4);
  int c4 = (int)(i - (long long)e * D4) << 2;  // channel base (Cch % 4 == 0)
  int h  = c4 / Cch;
  int s = (int)ei[e], d = (int)ei[E + e];
  float alpha = EXPA[(size_t)e * H + h] / (DEN[(size_t)d * H + h] + 1e-16f);
  const float4 v = *(const float4*)(Vb + (size_t)s * D + c4);
  float* o = OUTACC + (size_t)d * D + c4;
  atomicAdd(o + 0, alpha * v.x);
  atomicAdd(o + 1, alpha * v.y);
  atomicAdd(o + 2, alpha * v.z);
  atomicAdd(o + 3, alpha * v.w);
}

__global__ void relu_cast_kernel(const float* __restrict__ in,
                                 unsigned short* __restrict__ XB, int count) {
  int i = blockIdx.x * blockDim.x + threadIdx.x;
  if (i >= count) return;
  float v = in[i];
  XB[i] = f2bf(v > 0.0f ? v : 0.0f);
}

__global__ void final_copy_kernel(const float* __restrict__ in,
                                  float* __restrict__ out, int count) {
  int i = blockIdx.x * blockDim.x + threadIdx.x;
  if (i >= count) return;
  float v = in[i];
  out[i] = v;          // tuple element 0
  out[i + count] = v;  // tuple element 1 (reference returns (x, x))
}

// ------------------------------------------------- launch
extern "C" void kernel_launch(void* const* d_in, const int* in_sizes, int n_in,
                              void* d_out, int out_size, void* d_ws, size_t ws_size,
                              hipStream_t stream) {
  (void)in_sizes; (void)n_in; (void)out_size; (void)ws_size;

  const float*     feat = (const float*)d_in[0];
  const long long* ei   = (const long long*)d_in[1];   // [2, E] int64, row 0 = src, row 1 = dst
  const float*     pe   = (const float*)d_in[2];

  const int N = NN, E = NE;

  // ---- workspace carve-out (all 256B aligned) ----
  char* p = (char*)d_ws;
  auto take = [&](size_t bytes) {
    char* r = p;
    p += (bytes + 255) & ~(size_t)255;
    return r;
  };
  unsigned short* XB  = (unsigned short*)take((size_t)N * 256 * 2); // node feats, bf16, max D=256
  unsigned short* WBq = (unsigned short*)take((size_t)256 * 256 * 2);
  unsigned short* WBk = (unsigned short*)take((size_t)256 * 256 * 2);
  unsigned short* WBv = (unsigned short*)take((size_t)256 * 256 * 2);
  unsigned short* WBr = (unsigned short*)take((size_t)256 * 256 * 2);
  float* Q      = (float*)take((size_t)N * 256 * 4);
  float* Kb     = (float*)take((size_t)N * 256 * 4);
  float* Vb     = (float*)take((size_t)N * 256 * 4);
  float* OUTACC = (float*)take((size_t)N * 256 * 4);
  float* LOG    = (float*)take((size_t)E * 8 * 4);
  float* EXPA   = (float*)take((size_t)E * 8 * 4);
  unsigned int* MMAX = (unsigned int*)take((size_t)N * 8 * 4);
  float*        DEN  = (float*)take((size_t)N * 8 * 4);

  // ---- stage 0: concat(feature, pe) -> bf16 ----
  {
    int tot = N * 128;
    concat_cast_kernel<<<(tot + 255) / 256, 256, 0, stream>>>(feat, pe, XB, N, 120, 8);
  }

  const int layerK[3] = {128, 256, 256};   // input dim
  const int layerH[3] = {8, 8, 1};         // heads
  const int layerC[3] = {32, 32, 128};     // channels per head

  for (int li = 0; li < 3; ++li) {
    const int K = layerK[li], H = layerH[li], Cch = layerC[li], D = H * Cch;
    // param leaves in setup_inputs insertion order: Wq,bq,Wk,bk,Wv,bv,Wr,br
    const float* Wq = (const float*)d_in[3 + 8 * li + 0];
    const float* bq = (const float*)d_in[3 + 8 * li + 1];
    const float* Wk = (const float*)d_in[3 + 8 * li + 2];
    const float* bk = (const float*)d_in[3 + 8 * li + 3];
    const float* Wv = (const float*)d_in[3 + 8 * li + 4];
    const float* bv = (const float*)d_in[3 + 8 * li + 5];
    const float* Wr = (const float*)d_in[3 + 8 * li + 6];
    const float* br = (const float*)d_in[3 + 8 * li + 7];

    // weights -> bf16
    const int wcnt = K * D;
    const int wg = (wcnt + 255) / 256;
    cast_bf16_kernel<<<wg, 256, 0, stream>>>(Wq, WBq, wcnt);
    cast_bf16_kernel<<<wg, 256, 0, stream>>>(Wk, WBk, wcnt);
    cast_bf16_kernel<<<wg, 256, 0, stream>>>(Wv, WBv, wcnt);
    cast_bf16_kernel<<<wg, 256, 0, stream>>>(Wr, WBr, wcnt);

    // projections (WMMA); R lands directly in OUTACC so aggregation adds onto it
    const int waves  = (N / 16) * (D / 64);  // 16x64 strip per wave
    const int blocks = (waves + 7) / 8;      // 8 waves per 256-thread block
    if (li == 0) {
      gemm_bf16_wmma_kernel<128, 256><<<blocks, 256, 0, stream>>>(XB, WBq, bq, Q,      N);
      gemm_bf16_wmma_kernel<128, 256><<<blocks, 256, 0, stream>>>(XB, WBk, bk, Kb,     N);
      gemm_bf16_wmma_kernel<128, 256><<<blocks, 256, 0, stream>>>(XB, WBv, bv, Vb,     N);
      gemm_bf16_wmma_kernel<128, 256><<<blocks, 256, 0, stream>>>(XB, WBr, br, OUTACC, N);
    } else if (li == 1) {
      gemm_bf16_wmma_kernel<256, 256><<<blocks, 256, 0, stream>>>(XB, WBq, bq, Q,      N);
      gemm_bf16_wmma_kernel<256, 256><<<blocks, 256, 0, stream>>>(XB, WBk, bk, Kb,     N);
      gemm_bf16_wmma_kernel<256, 256><<<blocks, 256, 0, stream>>>(XB, WBv, bv, Vb,     N);
      gemm_bf16_wmma_kernel<256, 256><<<blocks, 256, 0, stream>>>(XB, WBr, br, OUTACC, N);
    } else {
      gemm_bf16_wmma_kernel<256, 128><<<blocks, 256, 0, stream>>>(XB, WBq, bq, Q,      N);
      gemm_bf16_wmma_kernel<256, 128><<<blocks, 256, 0, stream>>>(XB, WBk, bk, Kb,     N);
      gemm_bf16_wmma_kernel<256, 128><<<blocks, 256, 0, stream>>>(XB, WBv, bv, Vb,     N);
      gemm_bf16_wmma_kernel<256, 128><<<blocks, 256, 0, stream>>>(XB, WBr, br, OUTACC, N);
    }

    // segment softmax + aggregation
    const int nh = N * H;
    init_nh_kernel<<<(nh + 255) / 256, 256, 0, stream>>>(MMAX, DEN, nh);

    const float scale = 1.0f / sqrtf((float)Cch);
    const int eblocks = (E + 7) / 8;         // 8 edges (waves) per block
    edge_logits_max_kernel<<<eblocks, 256, 0, stream>>>(ei, Q, Kb, LOG, MMAX,
                                                        E, H, Cch, D, scale);
    const int eh = E * H;
    edge_exp_sum_kernel<<<(eh + 255) / 256, 256, 0, stream>>>(ei, LOG, MMAX, EXPA, DEN, E, H);

    const long long ed4 = (long long)E * (D >> 2);
    edge_aggregate_kernel<<<(int)((ed4 + 255) / 256), 256, 0, stream>>>(ei, Vb, EXPA, DEN,
                                                                        OUTACC, E, H, Cch, D);

    if (li < 2) {
      const int nd = N * D;
      relu_cast_kernel<<<(nd + 255) / 256, 256, 0, stream>>>(OUTACC, XB, nd);
    } else {
      const int cnt = N * D;  // 50000 * 128
      final_copy_kernel<<<(cnt + 255) / 256, 256, 0, stream>>>(OUTACC, (float*)d_out, cnt);
    }
  }
}
